// EdgeAttentionLayerFast_36197984370742
// MI455X (gfx1250) — matrix operand order repaired and checked by reading, not compiled, not run on hardware
//
#include <hip/hip_runtime.h>
#include <hip/hip_bf16.h>
#include <stdint.h>

// ---------------- problem constants (fixed by reference) ----------------
#define E_EDGES 262144
#define DM      128
#define NHEADS  4
#define DH      32
#define NNODES  65536
#define LN_EPS  1e-5f

typedef __attribute__((ext_vector_type(16))) __bf16          v16bf;
typedef __attribute__((ext_vector_type(8)))  float           v8f;
typedef __attribute__((ext_vector_type(16))) unsigned short  v16us;

// float -> bf16, round-to-nearest-even
__device__ __forceinline__ unsigned short f2bf(float f) {
  unsigned int u = __float_as_uint(f);
  unsigned int r = u + 0x7FFFu + ((u >> 16) & 1u);
  return (unsigned short)(r >> 16);
}
// order-preserving float<->uint encoding for atomicMax-based segment-max
__device__ __forceinline__ unsigned int encOrd(float f) {
  unsigned int u = __float_as_uint(f);
  return (f >= 0.0f) ? (u | 0x80000000u) : ~u;
}
__device__ __forceinline__ float decOrd(unsigned int u) {
  unsigned int b = (u & 0x80000000u) ? (u & 0x7FFFFFFFu) : ~u;
  return __uint_as_float(b);
}

// A-tile (16x32 bf16) loader from an fp32 row: ISA 16-bit A layout.
// lane: M = lane&15, half = lane>>4; elem j<8 -> K = kt*32 + 8*half + j,
// elem j>=8 -> K = kt*32 + 16 + 8*half + (j-8).
__device__ __forceinline__ v16bf load_a_f32row(const float* __restrict__ row, int kt, int half) {
  v16us u;
  const float* g0 = row + kt * 32 + 8 * half;
  const float* g1 = row + kt * 32 + 16 + 8 * half;
#pragma unroll
  for (int j = 0; j < 8; ++j) { u[j] = f2bf(g0[j]); u[8 + j] = f2bf(g1[j]); }
  return __builtin_bit_cast(v16bf, u);
}

// B-tile (32x16 bf16) loader from transposed bf16 weights (WT[n][k]).
// lane: N = lane&15; elem j<8 -> K = kt*32 + 8*half + j, elem j>=8 -> K = kt*32 + 16 + 8*half + (j-8).
__device__ __forceinline__ v16bf load_b_wt(const unsigned short* __restrict__ WT,
                                           int ncol, int kt, int half) {
  const unsigned short* p = WT + (size_t)ncol * DM + kt * 32 + 8 * half;
  v16us u;
#pragma unroll
  for (int j = 0; j < 8; ++j) { u[j] = p[j]; u[8 + j] = p[16 + j]; }
  return __builtin_bit_cast(v16bf, u);
}

// ---------------- prep: W (row-major f32) -> W^T (bf16) ----------------
__global__ void k_prep_weights(const float* __restrict__ Wq, const float* __restrict__ Wk,
                               const float* __restrict__ Wv, const float* __restrict__ Wo,
                               unsigned short* __restrict__ WTq, unsigned short* __restrict__ WTk,
                               unsigned short* __restrict__ WTv, unsigned short* __restrict__ WTo) {
  int t = blockIdx.x * blockDim.x + threadIdx.x;   // DM*DM
  int n = t >> 7, k = t & 127;
  WTq[n * DM + k] = f2bf(Wq[k * DM + n]);
  WTk[n * DM + k] = f2bf(Wk[k * DM + n]);
  WTv[n * DM + k] = f2bf(Wv[k * DM + n]);
  WTo[n * DM + k] = f2bf(Wo[k * DM + n]);
}

// ---------------- init node-level accumulators ----------------
__global__ void k_init(float* __restrict__ num, unsigned int* __restrict__ mEnc,
                       float* __restrict__ denom) {
  int t = blockIdx.x * blockDim.x + threadIdx.x;   // NNODES*DM
  num[t] = 0.0f;
  if (t < NNODES * NHEADS) { mEnc[t] = 0u; denom[t] = 0.0f; }
}

// ---------------- K1: fused QKV GEMM + per-head scores + segment max ----------------
__global__ void __launch_bounds__(256)
k_qkv_scores(const float* __restrict__ X, const long long* __restrict__ eidx,
             const unsigned short* __restrict__ WTq, const unsigned short* __restrict__ WTk,
             const unsigned short* __restrict__ WTv,
             const float* __restrict__ bq, const float* __restrict__ bk,
             const float* __restrict__ bv,
             float* __restrict__ Vbuf, float* __restrict__ scores,
             unsigned int* __restrict__ mEnc) {
  const int  wave = (blockIdx.x * blockDim.x + threadIdx.x) >> 5;
  const int  lane = threadIdx.x & 31;
  const int  half = lane >> 4;
  const int  mrow = lane & 15;
  const long long e0 = (long long)wave * 16;

  const float* row = X + (e0 + mrow) * DM;
  __builtin_prefetch(row + 16 * DM, 0, 0);         // global_prefetch next tile's rows

  v16bf a[4];
#pragma unroll
  for (int kt = 0; kt < 4; ++kt) a[kt] = load_a_f32row(row, kt, half);

  float headacc[NHEADS][8] = {};

#pragma unroll
  for (int nt = 0; nt < 8; ++nt) {
    v8f q = {}; v8f k = {}; v8f v = {};
#pragma unroll
    for (int kt = 0; kt < 4; ++kt) {
      v16bf btq = load_b_wt(WTq, nt * 16 + mrow, kt, half);
      q = __builtin_amdgcn_wmma_f32_16x16x32_bf16(false, a[kt], false, btq, (short)0, q, false, false);
      v16bf btk = load_b_wt(WTk, nt * 16 + mrow, kt, half);
      k = __builtin_amdgcn_wmma_f32_16x16x32_bf16(false, a[kt], false, btk, (short)0, k, false, false);
      v16bf btv = load_b_wt(WTv, nt * 16 + mrow, kt, half);
      v = __builtin_amdgcn_wmma_f32_16x16x32_bf16(false, a[kt], false, btv, (short)0, v, false, false);
    }
    const int col = nt * 16 + mrow;                // C/D layout: N = lane&15
    const float qb = bq[col], kb = bk[col], vb = bv[col];
#pragma unroll
    for (int r = 0; r < 8; ++r) {                  // row M = r + 8*half
      float qv = q[r] + qb, kv = k[r] + kb;
      headacc[nt >> 1][r] += qv * kv;              // head = col/32 = nt>>1
      Vbuf[(e0 + r + 8 * half) * DM + col] = v[r] + vb;
    }
  }

  const float scale = 0.17677669529663688f;        // 1/sqrt(Dh)
#pragma unroll
  for (int h = 0; h < NHEADS; ++h)
#pragma unroll
    for (int r = 0; r < 8; ++r) {
      float s = headacc[h][r];                     // reduce over N (lanes 0..15 of each half)
      s += __shfl_xor(s, 1); s += __shfl_xor(s, 2);
      s += __shfl_xor(s, 4); s += __shfl_xor(s, 8);
      headacc[h][r] = s * scale;
    }

  if (mrow < 8) {                                  // lane (r, half) owns row r+8*half
    const int r = mrow;
    const long long e = e0 + r + 8 * half;
    const int sn = (int)eidx[e];
    const int dn = (int)eidx[E_EDGES + e];
#pragma unroll
    for (int h = 0; h < NHEADS; ++h) {
      float s = headacc[h][r];
      scores[e * NHEADS + h] = s;
      unsigned int es = encOrd(s);
      atomicMax(&mEnc[sn * NHEADS + h], es);
      atomicMax(&mEnc[dn * NHEADS + h], es);
    }
  }
}

// ---------------- K2: scatter exp(s-m)*V into node numerator / denom ----------------
__global__ void __launch_bounds__(256)
k_scatter(const float* __restrict__ Vbuf, const float* __restrict__ scores,
          const long long* __restrict__ eidx, const unsigned int* __restrict__ mEnc,
          float* __restrict__ num, float* __restrict__ denom) {
  const long long gid = (long long)blockIdx.x * blockDim.x + threadIdx.x;  // E*DM
  const int d = (int)(gid & 127);
  const long long e = gid >> 7;
  const int h = d >> 5;
  const float v = Vbuf[gid];
  const float s = scores[e * NHEADS + h];
  const int n0 = (int)eidx[e];
  const int n1 = (int)eidx[E_EDGES + e];
#pragma unroll
  for (int i = 0; i < 2; ++i) {
    const int n = (i == 0) ? n0 : n1;
    const float m = decOrd(mEnc[n * NHEADS + h]);
    const float w = __expf(s - m);
    atomicAdd(&num[(size_t)n * DM + d], w * v);    // L2-resident (35 MB of node state)
    if ((d & 31) == 0) atomicAdd(&denom[n * NHEADS + h], w);
  }
}

// ---------------- K3: per-node divide numerator by softmax denominator ----------------
__global__ void k_nodediv(float* __restrict__ num, const float* __restrict__ denom) {
  int gid = blockIdx.x * blockDim.x + threadIdx.x; // NNODES*DM
  int n = gid >> 7, h = (gid & 127) >> 5;
  num[gid] = num[gid] / denom[n * NHEADS + h];
}

// ---------------- K4: gather msg, GEMM Wo (WMMA), residual + LayerNorm ----------------
__global__ void __launch_bounds__(256)
k_out(const float* __restrict__ X, const long long* __restrict__ eidx,
      const float* __restrict__ nodemsg, const unsigned short* __restrict__ WTo,
      const float* __restrict__ bo, const float* __restrict__ gamma,
      const float* __restrict__ beta, float* __restrict__ out) {
  const int  wave = (blockIdx.x * blockDim.x + threadIdx.x) >> 5;
  const int  lane = threadIdx.x & 31;
  const int  half = lane >> 4;
  const int  mrow = lane & 15;
  const long long e0 = (long long)wave * 16;

  const long long e = e0 + mrow;                   // this lane's A row
  const float* sp = nodemsg + (size_t)eidx[e] * DM;
  const float* dp = nodemsg + (size_t)eidx[E_EDGES + e] * DM;

  v16bf a[4];
#pragma unroll
  for (int kt = 0; kt < 4; ++kt) {
    v16us u;
    const int kb = kt * 32;
#pragma unroll
    for (int j = 0; j < 8; ++j) {
      int i0 = kb + 8 * half + j;
      int i1 = kb + 16 + 8 * half + j;
      u[j]     = f2bf(0.5f * (sp[i0] + dp[i0]));
      u[8 + j] = f2bf(0.5f * (sp[i1] + dp[i1]));
    }
    a[kt] = __builtin_bit_cast(v16bf, u);
  }

  float xv[8][8];
  float sum[8] = {}, sq[8] = {};
#pragma unroll
  for (int nt = 0; nt < 8; ++nt) {
    v8f c = {};
#pragma unroll
    for (int kt = 0; kt < 4; ++kt) {
      v16bf b = load_b_wt(WTo, nt * 16 + mrow, kt, half);
      c = __builtin_amdgcn_wmma_f32_16x16x32_bf16(false, a[kt], false, b, (short)0, c, false, false);
    }
    const int col = nt * 16 + mrow;
    const float bb = bo[col];
#pragma unroll
    for (int r = 0; r < 8; ++r) {
      float x = c[r] + bb + X[(e0 + r + 8 * half) * DM + col];  // residual
      xv[nt][r] = x;
      sum[r] += x;
      sq[r]  += x * x;
    }
  }
  float mu[8], inv[8];
#pragma unroll
  for (int r = 0; r < 8; ++r) {
    float s1 = sum[r], s2 = sq[r];                 // reduce across cols (16 lanes/half)
    s1 += __shfl_xor(s1, 1); s1 += __shfl_xor(s1, 2);
    s1 += __shfl_xor(s1, 4); s1 += __shfl_xor(s1, 8);
    s2 += __shfl_xor(s2, 1); s2 += __shfl_xor(s2, 2);
    s2 += __shfl_xor(s2, 4); s2 += __shfl_xor(s2, 8);
    float m  = s1 * (1.0f / DM);
    float va = s2 * (1.0f / DM) - m * m;
    mu[r] = m;
    inv[r] = rsqrtf(va + LN_EPS);
  }
#pragma unroll
  for (int nt = 0; nt < 8; ++nt) {
    const int col = nt * 16 + mrow;
    const float g = gamma[col], bt = beta[col];
#pragma unroll
    for (int r = 0; r < 8; ++r)
      out[(e0 + r + 8 * half) * DM + col] = (xv[nt][r] - mu[r]) * inv[r] * g + bt;
  }
}

// ---------------- host launch ----------------
extern "C" void kernel_launch(void* const* d_in, const int* in_sizes, int n_in,
                              void* d_out, int out_size, void* d_ws, size_t ws_size,
                              hipStream_t stream) {
  (void)in_sizes; (void)n_in; (void)out_size; (void)ws_size;
  const float*     X    = (const float*)d_in[0];
  const long long* eidx = (const long long*)d_in[1];   // (2,E) int64
  const float* Wq = (const float*)d_in[3];
  const float* bq = (const float*)d_in[4];
  const float* Wk = (const float*)d_in[5];
  const float* bk = (const float*)d_in[6];
  const float* Wv = (const float*)d_in[7];
  const float* bv = (const float*)d_in[8];
  const float* Wo = (const float*)d_in[9];
  const float* bo = (const float*)d_in[10];
  const float* gamma = (const float*)d_in[11];
  const float* beta  = (const float*)d_in[12];
  float* out = (float*)d_out;

  // workspace layout (~174 MB)
  char* ws = (char*)d_ws;
  size_t off = 0;
  float*          Vbuf   = (float*)(ws + off); off += (size_t)E_EDGES * DM * 4;
  float*          scores = (float*)(ws + off); off += (size_t)E_EDGES * NHEADS * 4;
  unsigned int*   mEnc   = (unsigned int*)(ws + off); off += (size_t)NNODES * NHEADS * 4;
  float*          denom  = (float*)(ws + off); off += (size_t)NNODES * NHEADS * 4;
  float*          num    = (float*)(ws + off); off += (size_t)NNODES * DM * 4;
  unsigned short* WTq    = (unsigned short*)(ws + off); off += (size_t)DM * DM * 2;
  unsigned short* WTk    = (unsigned short*)(ws + off); off += (size_t)DM * DM * 2;
  unsigned short* WTv    = (unsigned short*)(ws + off); off += (size_t)DM * DM * 2;
  unsigned short* WTo    = (unsigned short*)(ws + off); off += (size_t)DM * DM * 2;

  k_prep_weights<<<(DM * DM) / 256, 256, 0, stream>>>(Wq, Wk, Wv, Wo, WTq, WTk, WTv, WTo);
  k_init<<<(NNODES * DM) / 256, 256, 0, stream>>>(num, mEnc, denom);
  // one wave per 16-edge tile, 8 waves per 256-thread block
  k_qkv_scores<<<E_EDGES / 128, 256, 0, stream>>>(X, eidx, WTq, WTk, WTv, bq, bk, bv,
                                                  Vbuf, scores, mEnc);
  k_scatter<<<(unsigned)(((long long)E_EDGES * DM) / 256), 256, 0, stream>>>(
      Vbuf, scores, eidx, mEnc, num, denom);
  k_nodediv<<<(NNODES * DM) / 256, 256, 0, stream>>>(num, denom);
  k_out<<<E_EDGES / 128, 256, 0, stream>>>(X, eidx, num, WTo, bo, gamma, beta, out);
}
